// Embedder_20959440405113
// MI455X (gfx1250) — compile-verified, matrix operand order checked by reading, not run on hardware
//
#include <hip/hip_runtime.h>
#include <hip/hip_fp16.h>
#include <math.h>

typedef __attribute__((ext_vector_type(16))) _Float16 v16h;
typedef __attribute__((ext_vector_type(8)))  float    v8f;

#define BATCH   64
#define SEQ     8192
#define DIM     64
#define NBINS   256
#define CHUNK   32                  // K of one WMMA f16 op
#define NCHUNKS (SEQ / CHUNK)

// One block per batch row. 256 threads = 8 waves (wave32).
// Wave w owns bins [32w, 32w+32): 2 bin-tiles (M) x 4 d-tiles (N) of 16x16 f32 acc.
// sums[256,64] = OneHot[256xS] x Vals[Sx64], accumulated in WMMA f32 registers.
__global__ __launch_bounds__(256)
void embedder_onehot_wmma(const float* __restrict__ T,
                          const int*   __restrict__ X,
                          const float* __restrict__ embX,
                          const float* __restrict__ embW,
                          float*       __restrict__ out)
{
    __shared__ int s_bin[2][CHUNK];
    // B-matrix staging (double buffered): [buf][d_tile][lane][2r+half]
    __shared__ __align__(32) _Float16 s_B[2][4][32][16];
    __shared__ float s_cnt[NBINS];

    const int b    = blockIdx.x;
    const int tid  = threadIdx.x;
    const int lane = tid & 31;
    const int wave = tid >> 5;
    const int binbase = wave * 32;

    for (int i = tid; i < NBINS; i += 256) s_cnt[i] = 0.0f;
    __syncthreads();

    v8f zero = {};
    v8f acc[2][4];
    #pragma unroll
    for (int tm = 0; tm < 2; ++tm)
        #pragma unroll
        for (int td = 0; td < 4; ++td) acc[tm][td] = zero;

    // per-thread gather role: token k = tid/8, dims [8*(tid%8), +8)
    const int gk    = tid >> 3;
    const int dbase = (tid & 7) * 8;
    // inverse of A/B K-layout: k -> (r, half, khalf)
    const int g_hi    = (gk >= 16) ? 4 : 0;
    const int g_kp    = gk & 15;
    const int g_khalf = g_kp >> 3;
    const int g_rem   = g_kp & 7;
    const int g_r     = g_hi + (g_rem >> 1);
    const int g_half  = g_rem & 1;

    const int M     = lane & 15;
    const int khalf = lane >> 4;

    // ---- pipeline prologue: load chunk 0 into registers ----
    int   binval = -1;
    float w      = 0.0f;
    float vals[8];
    {
        if (tid < CHUNK) {
            float t = T[(size_t)b * SEQ + tid];
            binval  = (t < 256.0f) ? (int)t : -1;   // invalid -> matches no one-hot row
        }
        int id = X[(size_t)b * SEQ + gk];
        w = __expf(embW[id]);
        const float* row = embX + (size_t)id * DIM + dbase;
        #pragma unroll
        for (int j = 0; j < 8; ++j) vals[j] = row[j];
    }

    #pragma unroll 1
    for (int i = 0; i < NCHUNKS; ++i) {
        const int p = i & 1;

        // ---------- stage registers -> LDS buffer p ----------
        if (tid < CHUNK) {
            s_bin[p][tid] = binval;
            if (binval >= 0) atomicAdd(&s_cnt[binval], 1.0f);   // ds_add_f32
        }
        #pragma unroll
        for (int j = 0; j < 8; ++j) {
            int d = dbase + j;
            s_B[p][d >> 4][g_khalf * 16 + (d & 15)][2 * g_r + g_half] =
                (_Float16)(w * vals[j]);
        }
        __syncthreads();   // single barrier per chunk (double-buffered)

        // ---------- issue chunk i+1 global gathers (hidden under WMMA) ----------
        if (i + 1 < NCHUNKS) {
            const int s0 = (i + 1) * CHUNK;
            if (tid < CHUNK) {
                float t = T[(size_t)b * SEQ + s0 + tid];
                binval  = (t < 256.0f) ? (int)t : -1;
            }
            int id = X[(size_t)b * SEQ + s0 + gk];
            w = __expf(embW[id]);
            const float* row = embX + (size_t)id * DIM + dbase;
            #pragma unroll
            for (int j = 0; j < 8; ++j) vals[j] = row[j];
        }

        // ---------- one-hot A build + WMMA (EXEC all ones) ----------
        v16h a[2];
        #pragma unroll
        for (int r = 0; r < 8; ++r) {
            int kb = ((r < 4) ? 0 : 16) + khalf * 8 + (r & 3) * 2;
            int b0 = s_bin[p][kb], b1 = s_bin[p][kb + 1];
            #pragma unroll
            for (int tm = 0; tm < 2; ++tm) {
                int mybin = binbase + tm * 16 + M;
                a[tm][2 * r]     = (b0 == mybin) ? (_Float16)1.0f : (_Float16)0.0f;
                a[tm][2 * r + 1] = (b1 == mybin) ? (_Float16)1.0f : (_Float16)0.0f;
            }
        }
        // preload all 4 B tiles so the 8 WMMAs can issue without ds waits between
        v16h bm[4];
        #pragma unroll
        for (int td = 0; td < 4; ++td)
            bm[td] = *(const v16h*)&s_B[p][td][lane][0];   // 32B contiguous LDS read
        #pragma unroll
        for (int td = 0; td < 4; ++td) {
            #pragma unroll
            for (int tm = 0; tm < 2; ++tm) {
                acc[tm][td] = __builtin_amdgcn_wmma_f32_16x16x32_f16(
                    false, a[tm], false, bm[td], (short)0, acc[tm][td], false, false);
            }
        }
        // no second barrier: next iteration writes the other buffer
    }

    __syncthreads();   // counts complete before finalize reads

    // ---------- finalize: D layout -> out[b, bin, d] / (count + 1e-6) ----------
    const int M8 = (lane >> 4) * 8;
    const int N  = lane & 15;
    #pragma unroll
    for (int tm = 0; tm < 2; ++tm) {
        #pragma unroll
        for (int v = 0; v < 8; ++v) {
            int bin   = binbase + tm * 16 + M8 + v;
            float inv = 1.0f / (s_cnt[bin] + 1e-6f);
            #pragma unroll
            for (int td = 0; td < 4; ++td) {
                int d = td * 16 + N;
                out[((size_t)b * NBINS + bin) * DIM + d] = acc[tm][td][v] * inv;
            }
        }
    }
}

extern "C" void kernel_launch(void* const* d_in, const int* in_sizes, int n_in,
                              void* d_out, int out_size, void* d_ws, size_t ws_size,
                              hipStream_t stream) {
    const float* T    = (const float*)d_in[0];   // [64, 8192]
    const int*   X    = (const int*)  d_in[1];   // [64, 8192]
    const float* embX = (const float*)d_in[2];   // [100001, 64]
    const float* embW = (const float*)d_in[3];   // [100001, 1]
    float*       out  = (float*)d_out;           // [64, 256, 64]
    (void)in_sizes; (void)n_in; (void)out_size; (void)d_ws; (void)ws_size;

    embedder_onehot_wmma<<<dim3(BATCH), dim3(256), 0, stream>>>(T, X, embX, embW, out);
}
